// RNNEncoder_87153476370513
// MI455X (gfx1250) — matrix operand order repaired and checked by reading, not compiled, not run on hardware
//
#include <hip/hip_runtime.h>

// ---------------------------------------------------------------------------
// RNN encoder for MI455X (gfx1250, wave32, WMMA) — persistent-kernel version.
//   h_{t+1} = tanh( x_t @ W_i2h^T + b + (keep*h_t) @ W_h2h^T ),  keep = 0.5
// Design:
//   * bf16 operands / f32 accum via v_wmma_f32_16x16x32_bf16, fused K=1280.
//   * Weights repacked once into WMMA B-operand lane order; each wave then
//     holds its entire 40-chunk B strip in 320 VGPRs for all 512 timesteps
//     (wave32 allows 1024 VGPRs/lane) -> zero weight traffic in steady state.
//   * Hidden state carried as bf16 pre-packed in WMMA A-operand lane order,
//     keep=0.5 (exact power of 2) folded into the stored value: h-chunks of
//     the K loop are pure load+wmma.
//   * ONE persistent kernel (64 blocks x 256 thr = 512 waves); timesteps are
//     separated by a device-scope software barrier in workspace (arrive
//     counter + generation counter, reset each launch by an init kernel).
// ---------------------------------------------------------------------------

typedef __bf16 bf16_t;
typedef __attribute__((ext_vector_type(16))) __bf16 v16bf;
typedef __attribute__((ext_vector_type(8)))  float  v8f;
typedef __attribute__((ext_vector_type(4)))  float  vf4;

#define S_LEN   512
#define BATCH   256
#define FDIM    256
#define HDIM    1024
#define KTOT    1280          /* FDIM + HDIM               */
#define KCH_X   8             /* 32-wide K chunks from x   */
#define KCH_H   32            /* 32-wide K chunks from h   */
#define KCH_ALL (KCH_X + KCH_H)
#define NTILES  64            /* HDIM / 16                 */
#define KEEP    0.5f          /* 1 - dropout_p             */
#define NBLOCKS 64
#define BLK_ELEMS 512         /* bf16 elements per packed 16x32 / 32x16 block */

// ---------------------------------------------------------------------------
// Pack Wcat^T = [W_i2h ; W_h2h]^T (K=1280, N=1024) into bf16 B-operand order:
// block(kc, nT): lane L -> column n = nT*16 + (L&15), element e ->
//                k = kc*32 + (e&7) + 8*(L>>4) + 16*(e>>3).
// ---------------------------------------------------------------------------
__global__ void __launch_bounds__(256)
prep_w_kernel(const float* __restrict__ Wi,   // [HDIM, FDIM] row-major
              const float* __restrict__ Wh,   // [HDIM, HDIM] row-major
              bf16_t* __restrict__ Wt)        // packed, KTOT*HDIM elements
{
    int idx = blockIdx.x * blockDim.x + threadIdx.x;
    if (idx >= KTOT * HDIM) return;
    int e    = idx & 15;
    int lane = (idx >> 4) & 31;
    int blk  = idx >> 9;            // kc*NTILES + nT
    int nT   = blk & (NTILES - 1);
    int kc   = blk >> 6;
    int n    = nT * 16 + (lane & 15);
    int k    = kc * 32 + (e & 7) + 8 * (lane >> 4) + 16 * (e >> 3);
    float w  = (k < FDIM) ? Wi[n * FDIM + k] : Wh[n * HDIM + (k - FDIM)];
    Wt[idx]  = (bf16_t)w;
}

// ---------------------------------------------------------------------------
// Broadcast init_hidden [H] into the packed bf16 A-layout hidden buffer,
// pre-scaled by keep, and reset the grid-barrier counters.
// A layout: block(mT, kc); lane L -> row m = mT*16+(L&15),
// element e -> k = kc*32 + e + 8*((e>>3) + (L>>4)).
// ---------------------------------------------------------------------------
__global__ void __launch_bounds__(256)
init_hbf_kernel(const float* __restrict__ h_init, bf16_t* __restrict__ hbf,
                unsigned* __restrict__ sync_cnt)
{
    int idx = blockIdx.x * blockDim.x + threadIdx.x;   // BATCH*HDIM packed slots
    if (idx < 2) sync_cnt[idx] = 0u;                   // [0]=arrive, [1]=generation
    if (idx >= BATCH * HDIM) return;
    int e    = idx & 15;
    int lane = (idx >> 4) & 31;
    int kc   = (idx >> 9) & (KCH_H - 1);
    int n    = kc * 32 + e + 8 * ((e >> 3) + (lane >> 4));
    hbf[idx] = (bf16_t)(KEEP * h_init[n]);
}

// ---------------------------------------------------------------------------
// Build one A operand (16x32 bf16) for this lane from an f32 row of x.
// p points at rowBase + kChunkBase + 8*half (16B aligned).
// ---------------------------------------------------------------------------
__device__ __forceinline__ v16bf load_a_x(const float* __restrict__ p)
{
    vf4 f0 = *(const vf4*)(p);
    vf4 f1 = *(const vf4*)(p + 4);
    vf4 f2 = *(const vf4*)(p + 16);
    vf4 f3 = *(const vf4*)(p + 20);
    v16bf a;
    a[0]  = (bf16_t)f0[0]; a[1]  = (bf16_t)f0[1];
    a[2]  = (bf16_t)f0[2]; a[3]  = (bf16_t)f0[3];
    a[4]  = (bf16_t)f1[0]; a[5]  = (bf16_t)f1[1];
    a[6]  = (bf16_t)f1[2]; a[7]  = (bf16_t)f1[3];
    a[8]  = (bf16_t)f2[0]; a[9]  = (bf16_t)f2[1];
    a[10] = (bf16_t)f2[2]; a[11] = (bf16_t)f2[3];
    a[12] = (bf16_t)f3[0]; a[13] = (bf16_t)f3[1];
    a[14] = (bf16_t)f3[2]; a[15] = (bf16_t)f3[3];
    return a;
}

// ---------------------------------------------------------------------------
// Persistent kernel: 64 blocks x 256 threads (8 waves/block, 512 waves).
// blockIdx -> (mPair 0..7, nGroup 0..7); wave w -> nT = nGroup*8 + w.
// Wave computes a 32(M) x 16(N) strip, holding its full B strip in VGPRs.
// ---------------------------------------------------------------------------
__global__ void __launch_bounds__(256)
rnn_persistent_kernel(const float*  __restrict__ x,     // [S, B, F]
                      const bf16_t* __restrict__ Wt,    // packed B layout
                      const float*  __restrict__ bias,  // [HDIM]
                      bf16_t*       __restrict__ hbA,   // packed A layout
                      bf16_t*       __restrict__ hbB,   // packed A layout
                      float*        __restrict__ outp,  // [BATCH, HDIM]
                      unsigned*     __restrict__ sync_cnt) // [0]=arrive [1]=gen
{
    const int lane  = threadIdx.x & 31;
    const int wave  = threadIdx.x >> 5;
    const int mPair = blockIdx.x >> 3;                 // 0..7
    const int nT    = (blockIdx.x & 7) * 8 + wave;     // 0..63
    const int half  = lane >> 4;
    const int lm    = lane & 15;
    const int mT0   = mPair * 2;
    const int row0  = mPair * 32 + lm;

    // ---- load the whole B strip (40 chunks x 8 VGPRs = 320 VGPRs) -------
    const bf16_t* wp = Wt + (size_t)nT * BLK_ELEMS + lane * 16;
    v16bf breg[KCH_ALL];
#pragma unroll
    for (int kc = 0; kc < KCH_ALL; ++kc)
        breg[kc] = *(const v16bf*)(wp + (size_t)kc * (NTILES * BLK_ELEMS));

    // ---- per-wave epilogue constants ------------------------------------
    const int   ncol = nT * 16 + lm;
    const float bn   = bias[ncol];
    const int   kcp  = ncol >> 5;
    const int   kl   = ncol & 31;
    const int   hh   = (kl >> 3) & 1;
    const int   ep   = (kl & 7) + 8 * (kl >> 4);
    const size_t scat0 = ((size_t)(mT0 * KCH_H + kcp) * 32 + 16 * hh) * 16 + ep;

    for (unsigned s = 0; s < S_LEN; ++s) {
        const float*  xs   = x + (size_t)s * BATCH * FDIM;
        const bf16_t* hsrc = (s & 1) ? hbB : hbA;
        bf16_t*       hdst = (s & 1) ? hbA : hbB;

        const float*  x0  = xs + row0 * FDIM;
        const float*  x1  = xs + (row0 + 16) * FDIM;
        const bf16_t* ha0 = hsrc + ((size_t)mT0 * KCH_H) * BLK_ELEMS + lane * 16;
        const bf16_t* ha1 = ha0 + (size_t)KCH_H * BLK_ELEMS;

        v8f acc0 = {0.f, 0.f, 0.f, 0.f, 0.f, 0.f, 0.f, 0.f};
        v8f acc1 = {0.f, 0.f, 0.f, 0.f, 0.f, 0.f, 0.f, 0.f};

        // x part: f32 load + cvt (co-executes with XDL wmma)
#pragma unroll
        for (int kc = 0; kc < KCH_X; ++kc) {
            const v16bf a0 = load_a_x(x0 + kc * 32 + 8 * half);
            const v16bf a1 = load_a_x(x1 + kc * 32 + 8 * half);
            acc0 = __builtin_amdgcn_wmma_f32_16x16x32_bf16(false, a0, false, breg[kc],
                                                           (short)0, acc0, false, false);
            acc1 = __builtin_amdgcn_wmma_f32_16x16x32_bf16(false, a1, false, breg[kc],
                                                           (short)0, acc1, false, false);
        }
        // h part: pure packed-bf16 load + wmma
#pragma unroll
        for (int kc = 0; kc < KCH_H; ++kc) {
            const v16bf a0 = *(const v16bf*)(ha0 + (size_t)kc * BLK_ELEMS);
            const v16bf a1 = *(const v16bf*)(ha1 + (size_t)kc * BLK_ELEMS);
            acc0 = __builtin_amdgcn_wmma_f32_16x16x32_bf16(false, a0, false, breg[KCH_X + kc],
                                                           (short)0, acc0, false, false);
            acc1 = __builtin_amdgcn_wmma_f32_16x16x32_bf16(false, a1, false, breg[KCH_X + kc],
                                                           (short)0, acc1, false, false);
        }

        // epilogue: bias + tanh
        if (s == S_LEN - 1) {
            float* d0 = outp + (size_t)(mPair * 32 + 8 * half) * HDIM + ncol;
            float* d1 = d0 + (size_t)16 * HDIM;
#pragma unroll
            for (int r = 0; r < 8; ++r) {
                d0[(size_t)r * HDIM] = tanhf(acc0[r] + bn);
                d1[(size_t)r * HDIM] = tanhf(acc1[r] + bn);
            }
            // last step: no barrier needed
        } else {
            // scatter keep*tanh into packed A layout for the next step
            bf16_t* b0 = hdst + scat0;
            bf16_t* b1 = b0 + (size_t)KCH_H * BLK_ELEMS;
#pragma unroll
            for (int r = 0; r < 8; ++r) {
                const int lr = 8 * half + r;
                b0[(size_t)lr * 16] = (bf16_t)(KEEP * tanhf(acc0[r] + bn));
                b1[(size_t)lr * 16] = (bf16_t)(KEEP * tanhf(acc1[r] + bn));
            }

            // ---- device-wide barrier between timesteps ------------------
            __threadfence();        // make this wave's h stores agent-visible
            __syncthreads();
            if (threadIdx.x == 0) {
                unsigned prev = __hip_atomic_fetch_add(&sync_cnt[0], 1u,
                                    __ATOMIC_ACQ_REL, __HIP_MEMORY_SCOPE_AGENT);
                if (prev == NBLOCKS - 1) {
                    __hip_atomic_store(&sync_cnt[0], 0u,
                                       __ATOMIC_RELAXED, __HIP_MEMORY_SCOPE_AGENT);
                    __hip_atomic_fetch_add(&sync_cnt[1], 1u,
                                       __ATOMIC_RELEASE, __HIP_MEMORY_SCOPE_AGENT);
                } else {
                    while (__hip_atomic_load(&sync_cnt[1], __ATOMIC_ACQUIRE,
                                             __HIP_MEMORY_SCOPE_AGENT) <= s)
                        __builtin_amdgcn_s_sleep(2);
                }
            }
            __syncthreads();
            __threadfence();        // invalidate stale lines before reading h
        }
    }
}

// ---------------------------------------------------------------------------
extern "C" void kernel_launch(void* const* d_in, const int* in_sizes, int n_in,
                              void* d_out, int out_size, void* d_ws, size_t ws_size,
                              hipStream_t stream)
{
    (void)in_sizes; (void)n_in; (void)out_size; (void)ws_size;

    const float* x      = (const float*)d_in[0];  // [S, B, F]
    const float* W_i2h  = (const float*)d_in[1];  // [H, F]
    const float* b_i2h  = (const float*)d_in[2];  // [H]
    const float* W_h2h  = (const float*)d_in[3];  // [H, H]
    const float* h_init = (const float*)d_in[4];  // [1, 1, H]
    float*       out    = (float*)d_out;          // [1, B, H]

    char*     ws   = (char*)d_ws;
    bf16_t*   Wt   = (bf16_t*)ws;                                   // 2.5 MiB
    bf16_t*   hbA  = (bf16_t*)(ws + (size_t)KTOT * HDIM * 2);       // 512 KiB
    bf16_t*   hbB  = (bf16_t*)(ws + (size_t)KTOT * HDIM * 2
                                  + (size_t)BATCH * HDIM * 2);      // 512 KiB
    unsigned* sync = (unsigned*)(ws + (size_t)KTOT * HDIM * 2
                                    + (size_t)BATCH * HDIM * 4);    // 2 x u32

    prep_w_kernel<<<(KTOT * HDIM) / 256, 256, 0, stream>>>(W_i2h, W_h2h, Wt);
    init_hbf_kernel<<<(BATCH * HDIM) / 256, 256, 0, stream>>>(h_init, hbA, sync);
    rnn_persistent_kernel<<<NBLOCKS, 256, 0, stream>>>(
        x, Wt, b_i2h, hbA, hbB, out, sync);
}
